// PeriodicDynamicGCN_51084341018991
// MI455X (gfx1250) — matrix operand order repaired and checked by reading, not compiled, not run on hardware
//
#include <hip/hip_runtime.h>
#include <hip/hip_bf16.h>

typedef __attribute__((ext_vector_type(16))) __bf16 v16bf;
typedef __attribute__((ext_vector_type(8)))  __bf16 v8bf;
typedef __attribute__((ext_vector_type(8)))  float  v8f;

#define B_  32
#define N_  1024
#define F_  128
#define H_  64
#define E_  16
#define P_  24

static __device__ __forceinline__ v8f wmma_bf16(v16bf a, v16bf b, v8f c) {
  return __builtin_amdgcn_wmma_f32_16x16x32_bf16(false, a, false, b, (short)0, c, false, false);
}

static __device__ __forceinline__ v16bf cat16(v8bf lo, v8bf hi) {
  return __builtin_shufflevector(lo, hi, 0,1,2,3,4,5,6,7,8,9,10,11,12,13,14,15);
}

// async copy: 16B per lane, global -> LDS, tracked by ASYNCcnt.
// INST_OFFSET is added to BOTH the LDS and the global address.
static __device__ __forceinline__ void async_b128(unsigned lds_off, const void* gaddr, int ioff) {
  unsigned long long ga = (unsigned long long)(uintptr_t)gaddr;
  switch (ioff) {
    case 0:  asm volatile("global_load_async_to_lds_b128 %0, %1, off"           :: "v"(lds_off), "v"(ga) : "memory"); break;
    case 16: asm volatile("global_load_async_to_lds_b128 %0, %1, off offset:16" :: "v"(lds_off), "v"(ga) : "memory"); break;
    case 32: asm volatile("global_load_async_to_lds_b128 %0, %1, off offset:32" :: "v"(lds_off), "v"(ga) : "memory"); break;
    default: asm volatile("global_load_async_to_lds_b128 %0, %1, off offset:48" :: "v"(lds_off), "v"(ga) : "memory"); break;
  }
}
static __device__ __forceinline__ void wait_async0() {
  asm volatile("s_wait_asynccnt 0x0" ::: "memory");
}

// ---------- prep kernels ----------

__global__ void k_env(const float* __restrict__ envf, const float* __restrict__ envW,
                      const float* __restrict__ envb, float* __restrict__ env_emb) {
  int i = blockIdx.x * blockDim.x + threadIdx.x;          // N*H = 65536
  int n = i >> 6, h = i & 63;
  float s = envb[h];
#pragma unroll
  for (int e = 0; e < E_; ++e) s += envf[n * E_ + e] * envW[e * H_ + h];
  env_emb[i] = s > 0.f ? s : 0.f;
}

__global__ void k_emb(const float* __restrict__ se, const float* __restrict__ te,
                      const float* __restrict__ env_emb,
                      __bf16* __restrict__ src_bf, __bf16* __restrict__ tgt_bf) {
  int i = blockIdx.x * blockDim.x + threadIdx.x;          // P*N*H
  float e = env_emb[i & (N_ * H_ - 1)];
  src_bf[i] = (__bf16)(se[i] + e);
  tgt_bf[i] = (__bf16)(te[i] + e);
}

__global__ void k_cvt(const float* __restrict__ x, __bf16* __restrict__ y) {
  int i = blockIdx.x * blockDim.x + threadIdx.x;          // B*N*F
  y[i] = (__bf16)x[i];
}

__global__ void k_wt(const float* __restrict__ w, const float* __restrict__ rw,
                     __bf16* __restrict__ wt, __bf16* __restrict__ rwt) {
  int i = blockIdx.x * blockDim.x + threadIdx.x;          // F*F = 16384
  int f = i >> 7, k = i & 127;
  wt[f * F_ + k]  = (__bf16)w[k * F_ + f];
  rwt[f * F_ + k] = (__bf16)rw[k * F_ + f];
}

// ---------- support = X @ W, stored transposed bf16 as supportT[b][f][n] ----------
__global__ __launch_bounds__(128) void k_support(const __bf16* __restrict__ in_bf,
                                                 const __bf16* __restrict__ weight_t,
                                                 __bf16* __restrict__ supportT) {
  int wave = (blockIdx.x * blockDim.x + threadIdx.x) >> 5;  // 2048 waves
  int lane = threadIdx.x & 31;
  int b  = wave >> 6;
  int n0 = (wave & 63) << 4;
  int cl = lane & 15, hl = lane >> 4;

  const __bf16* arow = in_bf + (size_t)(b * N_ + n0 + cl) * F_;
  v8f acc[8];
#pragma unroll
  for (int t = 0; t < 8; ++t) acc[t] = (v8f){0,0,0,0,0,0,0,0};

#pragma unroll
  for (int kk = 0; kk < 4; ++kk) {
    v16bf A = cat16(*(const v8bf*)(arow + kk * 32 + 16 * hl),
                    *(const v8bf*)(arow + kk * 32 + 8 + 16 * hl));
#pragma unroll
    for (int t = 0; t < 8; ++t) {
      v16bf Bt = *(const v16bf*)(weight_t + (t * 16 + cl) * F_ + kk * 32 + 16 * hl);
      acc[t] = wmma_bf16(A, Bt, acc[t]);
    }
  }
#pragma unroll
  for (int t = 0; t < 8; ++t) {
    v8bf s;
#pragma unroll
    for (int r = 0; r < 8; ++r) s[r] = (__bf16)acc[t][r];
    *(v8bf*)(supportT + ((size_t)b * F_ + t * 16 + cl) * N_ + n0 + 8 * hl) = s;
  }
}

// ---------- fused: scores -> relu*mask -> adj@support + bias + relu(residual), relu ----------
__global__ __launch_bounds__(128) void k_main(const __bf16* __restrict__ src_bf,
                                              const __bf16* __restrict__ tgt_bf,
                                              const __bf16* __restrict__ supportT,
                                              const __bf16* __restrict__ in_bf,
                                              const __bf16* __restrict__ resW_t,
                                              const float* __restrict__ static_adj,
                                              const float* __restrict__ bias,
                                              const float* __restrict__ res_b,
                                              const long long* __restrict__ cycle,
                                              float* __restrict__ out) {
  // double-buffered block-shared operand stage + wave-private transpose stage
  __shared__ __align__(32) __bf16 ldsSup[2][F_ * 32];   // [buf][f][m_local]  8KB each
  __shared__ __align__(32) __bf16 ldsTgt[2][32 * H_];   // [buf][m_local][h]  4KB each
  __shared__ float sS[4][16 * 33];

  int tid  = threadIdx.x;
  int wave = tid >> 5, lane = tid & 31;
  int tile = blockIdx.x * 4 + wave;                    // 2048 tiles (B * N/16); same b,p per block
  int b  = tile >> 6;
  int n0 = (tile & 63) << 4;
  int cl = lane & 15, hl = lane >> 4;
  int p  = (int)(cycle[b] % (long long)P_);
  float* ls = sS[wave];

  const __bf16* tbase = tgt_bf + (size_t)p * (N_ * H_);
  const __bf16* sup   = supportT + (size_t)b * (F_ * N_);

  // cooperative async prefetch of one 32-wide m chunk into LDS buffer `buf`
  auto prefetch = [&](int buf, int m0) {
    // supportT rows: thread t copies f=t, 64 bytes (m_local 0..31)
    const __bf16* gs = sup + (size_t)tid * N_ + m0;
    unsigned lsup = (unsigned)(uintptr_t)&ldsSup[buf][tid * 32];
    async_b128(lsup, gs, 0);
    async_b128(lsup, gs, 16);
    async_b128(lsup, gs, 32);
    async_b128(lsup, gs, 48);
    // tgt rows: thread t copies 32 bytes of row m_local = t>>2
    const __bf16* gt = tbase + (size_t)(m0 + (tid >> 2)) * H_ + (tid & 3) * 16;
    unsigned ltgt = (unsigned)(uintptr_t)&ldsTgt[buf][(tid >> 2) * H_ + (tid & 3) * 16];
    async_b128(ltgt, gt, 0);
    async_b128(ltgt, gt, 16);
  };

  // ---- residual = in @ res_W ----
  v8f racc[8];
#pragma unroll
  for (int t = 0; t < 8; ++t) racc[t] = (v8f){0,0,0,0,0,0,0,0};
  const __bf16* arow = in_bf + (size_t)(b * N_ + n0 + cl) * F_;
#pragma unroll
  for (int kk = 0; kk < 4; ++kk) {
    v16bf A = cat16(*(const v8bf*)(arow + kk * 32 + 16 * hl),
                    *(const v8bf*)(arow + kk * 32 + 8 + 16 * hl));
#pragma unroll
    for (int t = 0; t < 8; ++t) {
      v16bf Bt = *(const v16bf*)(resW_t + (t * 16 + cl) * F_ + kk * 32 + 16 * hl);
      racc[t] = wmma_bf16(A, Bt, racc[t]);
    }
  }

  // ---- A operand for scores: 16 src rows, K = H = 64 ----
  const __bf16* srow = src_bf + (size_t)(p * N_ + n0 + cl) * H_;
  v16bf A0 = cat16(*(const v8bf*)(srow + 16 * hl),      *(const v8bf*)(srow + 8 + 16 * hl));
  v16bf A1 = cat16(*(const v8bf*)(srow + 32 + 16 * hl), *(const v8bf*)(srow + 40 + 16 * hl));

  v8f acc[8];
#pragma unroll
  for (int t = 0; t < 8; ++t) acc[t] = (v8f){0,0,0,0,0,0,0,0};

  const float* mrow = static_adj + (size_t)(n0 + cl) * N_;

  prefetch(0, 0);
  for (int mc = 0; mc < 32; ++mc) {
    int m0 = mc * 32;
    int buf = mc & 1;
    wait_async0();          // my async copies for chunk mc are in LDS
    __syncthreads();        // everyone's are; previous buffer fully consumed
    if (mc + 1 < 32) prefetch(buf ^ 1, m0 + 32);   // overlap next chunk with compute

    const __bf16* Tg = &ldsTgt[buf][0];
    const __bf16* Sp = &ldsSup[buf][0];

    // scores tile 16x32 = src @ tgt^T  (B operand from LDS)
#pragma unroll
    for (int t2 = 0; t2 < 2; ++t2) {
      const __bf16* trow = Tg + (t2 * 16 + cl) * H_;
      v16bf Bt0 = *(const v16bf*)(trow + 16 * hl);
      v16bf Bt1 = *(const v16bf*)(trow + 32 + 16 * hl);
      v8f s = (v8f){0,0,0,0,0,0,0,0};
      s = wmma_bf16(A0, Bt0, s);
      s = wmma_bf16(A1, Bt1, s);
#pragma unroll
      for (int r = 0; r < 8; ++r)
        ls[(r + 8 * hl) * 33 + t2 * 16 + cl] = s[r];
    }
    // D-layout -> A-layout with relu * static_adj, convert to bf16
    v16bf A2;
#pragma unroll
    for (int j = 0; j < 8; ++j) {
      int k1 = 16 * hl + j, k2 = 16 * hl + 8 + j;
      float v1 = ls[cl * 33 + k1]; v1 = v1 > 0.f ? v1 : 0.f; v1 *= mrow[m0 + k1];
      float v2 = ls[cl * 33 + k2]; v2 = v2 > 0.f ? v2 : 0.f; v2 *= mrow[m0 + k2];
      A2[j]     = (__bf16)v1;
      A2[8 + j] = (__bf16)v2;
    }
    // accumulate adj_chunk @ support_chunk (B operand from LDS)
#pragma unroll
    for (int t = 0; t < 8; ++t) {
      v16bf Bs = *(const v16bf*)(Sp + (t * 16 + cl) * 32 + 16 * hl);
      acc[t] = wmma_bf16(A2, Bs, acc[t]);
    }
  }

  // ---- epilogue: relu(conv + bias + relu(res + res_b)) ----
#pragma unroll
  for (int t = 0; t < 8; ++t) {
    int f = t * 16 + cl;
    float bi = bias[f], rb = res_b[f];
#pragma unroll
    for (int r = 0; r < 8; ++r) {
      float rv = racc[t][r] + rb; rv = rv > 0.f ? rv : 0.f;
      float v  = acc[t][r] + bi + rv;
      out[(size_t)(b * N_ + n0 + r + 8 * hl) * F_ + f] = v > 0.f ? v : 0.f;
    }
  }
}

extern "C" void kernel_launch(void* const* d_in, const int* in_sizes, int n_in,
                              void* d_out, int out_size, void* d_ws, size_t ws_size,
                              hipStream_t stream) {
  (void)in_sizes; (void)n_in; (void)out_size; (void)ws_size;
  const float*     input_features = (const float*)d_in[0];
  const long long* cycle_indices  = (const long long*)d_in[1];
  const float*     weight         = (const float*)d_in[2];
  const float*     bias           = (const float*)d_in[3];
  const float*     src_emb        = (const float*)d_in[4];
  const float*     tgt_emb        = (const float*)d_in[5];
  const float*     env_W          = (const float*)d_in[6];
  const float*     env_b          = (const float*)d_in[7];
  const float*     res_W          = (const float*)d_in[8];
  const float*     res_b          = (const float*)d_in[9];
  const float*     static_adj     = (const float*)d_in[10];
  const float*     env_features   = (const float*)d_in[11];
  float* out = (float*)d_out;

  char* ws = (char*)d_ws;
  size_t off = 0;
  auto alloc = [&](size_t bytes) -> void* {
    void* p = ws + off;
    off += (bytes + 255) & ~(size_t)255;
    return p;
  };
  float*  env_emb  = (float*)alloc((size_t)N_ * H_ * 4);
  __bf16* src_bf   = (__bf16*)alloc((size_t)P_ * N_ * H_ * 2);
  __bf16* tgt_bf   = (__bf16*)alloc((size_t)P_ * N_ * H_ * 2);
  __bf16* in_bf    = (__bf16*)alloc((size_t)B_ * N_ * F_ * 2);
  __bf16* weight_t = (__bf16*)alloc((size_t)F_ * F_ * 2);
  __bf16* resW_t   = (__bf16*)alloc((size_t)F_ * F_ * 2);
  __bf16* supportT = (__bf16*)alloc((size_t)B_ * F_ * N_ * 2);

  k_env<<<(N_ * H_) / 256, 256, 0, stream>>>(env_features, env_W, env_b, env_emb);
  k_emb<<<(P_ * N_ * H_) / 256, 256, 0, stream>>>(src_emb, tgt_emb, env_emb, src_bf, tgt_bf);
  k_cvt<<<(B_ * N_ * F_) / 256, 256, 0, stream>>>(input_features, in_bf);
  k_wt<<<(F_ * F_) / 256, 256, 0, stream>>>(weight, res_W, weight_t, resW_t);
  k_support<<<(B_ * N_ / 16) / 4, 128, 0, stream>>>(in_bf, weight_t, supportT);
  k_main<<<(B_ * N_ / 16) / 4, 128, 0, stream>>>(src_bf, tgt_bf, supportT, in_bf, resW_t,
                                                 static_adj, bias, res_b, cycle_indices, out);
}
